// EfficientTransformerBlock_6365141532868
// MI455X (gfx1250) — compile-verified, matrix operand order checked by reading, not run on hardware
//
#include <hip/hip_runtime.h>
#include <math.h>

typedef __bf16 bf16;
typedef bf16  v16bf __attribute__((ext_vector_type(16)));
typedef float v8f   __attribute__((ext_vector_type(8)));

// ---------------------------------------------------------------------------
// WMMA helpers (CDNA5 gfx1250, wave32). D = A(16x32 bf16) x B(32x16 bf16) + C(f32)
// ---------------------------------------------------------------------------
__device__ __forceinline__ v8f wmma_bf16(v16bf a, v16bf b, v8f c) {
  return __builtin_amdgcn_wmma_f32_16x16x32_bf16(false, a, false, b, (short)0, c,
                                                 false, false);
}

__device__ __forceinline__ v8f acc_zero() {
  v8f a;
#pragma unroll
  for (int i = 0; i < 8; ++i) a[i] = 0.0f;
  return a;
}

// Async global->LDS copy, 16 bytes per call (CDNA5 GLOBAL_LOAD_ASYNC_TO_LDS_B128).
// lds_dst: pointer into __shared__ (low 32 bits of generic address == LDS offset).
__device__ __forceinline__ void async_cp16(void* lds_dst, const void* gsrc) {
  unsigned l = (unsigned)(uintptr_t)lds_dst;
  asm volatile("global_load_async_to_lds_b128 %0, %1, off"
               :
               : "v"(l), "v"(gsrc)
               : "memory");
}
__device__ __forceinline__ void async_wait0() {
  asm volatile("s_wait_asynccnt 0" ::: "memory");
}

// Row-major fragment: rows indexed by (row0 + lane&15), K contiguous.
__device__ __forceinline__ v16bf frag_ld(const bf16* base, int ld, int row0, int k0) {
  int lane = threadIdx.x & 31;
  const bf16* p = base + (size_t)(row0 + (lane & 15)) * ld + k0 + ((lane >> 4) << 3);
  v16bf f;
#pragma unroll
  for (int i = 0; i < 8; ++i) f[i] = p[i];
#pragma unroll
  for (int i = 0; i < 8; ++i) f[8 + i] = p[16 + i];
  return f;
}

// Transposed fragment: buffer stored [K][col] row-major but we need [col][K].
__device__ __forceinline__ v16bf frag_ld_t(const bf16* base, int ld, int col0, int k0) {
  int lane = threadIdx.x & 31;
  int c  = col0 + (lane & 15);
  int kb = k0 + ((lane >> 4) << 3);
  v16bf f;
#pragma unroll
  for (int i = 0; i < 8; ++i) f[i] = base[(size_t)(kb + i) * ld + c];
#pragma unroll
  for (int i = 0; i < 8; ++i) f[8 + i] = base[(size_t)(kb + 16 + i) * ld + c];
  return f;
}

#define TOK1 64
#define DPAD 72

// ---------------------------------------------------------------------------
// K0: zero accumulators + one-shot f32->bf16 weight conversion.
// ---------------------------------------------------------------------------
__global__ void k0_prep(const float* __restrict__ Wq, const float* __restrict__ Wk,
                        const float* __restrict__ Wv, const float* __restrict__ Wr,
                        const float* __restrict__ W1, const float* __restrict__ W2,
                        bf16* wq_b, bf16* wk_b, bf16* wv_b, bf16* wr_b, bf16* w1_b,
                        bf16* w2_b, float* ksum, float* ctx, int nk, int nc) {
  int i = blockIdx.x * blockDim.x + threadIdx.x;
  if (i < nk) ksum[i] = 0.0f;
  if (i < nc) ctx[i] = 0.0f;
  if (i < 4096) {
    wq_b[i] = (bf16)Wq[i];
    wk_b[i] = (bf16)Wk[i];
    wv_b[i] = (bf16)Wv[i];
    wr_b[i] = (bf16)Wr[i];
  }
  if (i < 16384) {
    w1_b[i] = (bf16)W1[i];
    w2_b[i] = (bf16)W2[i];
  }
}

// ---------------------------------------------------------------------------
// K1: LN1 + Q/K/V projections (WMMA) + q channel-softmax + k spatial exp
//     + partial context = kexp^T @ v  (atomic accumulate into ctx / ksum)
// block = 128 threads (4 waves), 64 tokens.
// ---------------------------------------------------------------------------
__global__ __launch_bounds__(128) void k1_proj(
    const float* __restrict__ x, const float* __restrict__ g1,
    const float* __restrict__ b1ln, const bf16* __restrict__ wq_b,
    const float* __restrict__ bq, const bf16* __restrict__ wk_b,
    const float* __restrict__ bk, const bf16* __restrict__ wv_b,
    const float* __restrict__ bv, bf16* __restrict__ qs_g,
    float* __restrict__ ksum, float* __restrict__ ctx, int N) {
  __shared__ __align__(16) float xs[TOK1][64];      // 16 KB
  __shared__ __align__(16) bf16  n1s[TOK1][DPAD];   // 9 KB
  __shared__ __align__(16) bf16  wgt[64][64];       // 8 KB (one weight at a time)
  __shared__ __align__(16) bf16  sbuf[TOK1][DPAD];  // 9 KB (q logits, then kexp)
  __shared__ __align__(16) bf16  vbuf[TOK1][DPAD];  // 9 KB
  __shared__ float biass[64];

  const int tid = threadIdx.x;
  const int wv  = tid >> 5;
  const int lane = tid & 31;
  const int tilesPerBatch = N / TOK1;
  const int b    = blockIdx.x / tilesPerBatch;
  const int tok0 = (blockIdx.x % tilesPerBatch) * TOK1;

  // ---- async load x tile (64 rows x 256B) ----
  for (int i = tid; i < TOK1 * 16; i += 128) {
    int tr = i >> 4, c = i & 15;
    async_cp16(&xs[tr][c * 4], x + ((size_t)(b * N + tok0 + tr) << 6) + c * 4);
  }
  async_wait0();
  __syncthreads();

  // ---- LN1 (one token per thread) ----
  if (tid < TOK1) {
    float mu = 0.f;
#pragma unroll
    for (int d = 0; d < 64; ++d) mu += xs[tid][d];
    mu *= (1.0f / 64.0f);
    float var = 0.f;
#pragma unroll
    for (int d = 0; d < 64; ++d) { float t = xs[tid][d] - mu; var += t * t; }
    var *= (1.0f / 64.0f);
    float inv = rsqrtf(var + 1e-5f);
#pragma unroll
    for (int d = 0; d < 64; ++d)
      n1s[tid][d] = (bf16)((xs[tid][d] - mu) * inv * g1[d] + b1ln[d]);
  }
  __syncthreads();

  // ---- three projections: Q, K, V ----
#pragma unroll 1
  for (int proj = 0; proj < 3; ++proj) {
    const bf16*  Wp = (proj == 0) ? wq_b : (proj == 1) ? wk_b : wv_b;
    const float* bp = (proj == 0) ? bq : (proj == 1) ? bk : bv;
    for (int i = tid; i < 512; i += 128)  // 4096 bf16 = 512 x 16B
      async_cp16(&wgt[0][0] + i * 8, Wp + i * 8);
    if (tid < 64) biass[tid] = bp[tid];
    async_wait0();
    __syncthreads();

    bf16* dst = (proj == 2) ? &vbuf[0][0] : &sbuf[0][0];
    const int mi = wv;  // wave -> M tile
#pragma unroll
    for (int nt = 0; nt < 4; ++nt) {
      v8f acc = acc_zero();
#pragma unroll
      for (int kt = 0; kt < 2; ++kt) {
        v16bf a  = frag_ld(&n1s[0][0], DPAD, mi * 16, kt * 32);
        v16bf bb = frag_ld(&wgt[0][0], 64, nt * 16, kt * 32);
        acc = wmma_bf16(a, bb, acc);
      }
      int col = nt * 16 + (lane & 15);
      int rb  = mi * 16 + ((lane >> 4) << 3);
#pragma unroll
      for (int r = 0; r < 8; ++r)
        dst[(size_t)(rb + r) * DPAD + col] = (bf16)(acc[r] + biass[col]);
    }
    __syncthreads();

    if (proj == 0) {
      // q: softmax over channels (axis -1), write bf16 to global qs
      if (tid < TOK1) {
        float mx = -1e30f;
#pragma unroll
        for (int d = 0; d < 64; ++d) mx = fmaxf(mx, (float)sbuf[tid][d]);
        float s = 0.f;
#pragma unroll
        for (int d = 0; d < 64; ++d) {
          float e = __expf((float)sbuf[tid][d] - mx);
          s += e;
          sbuf[tid][d] = (bf16)e;
        }
        float inv = 1.0f / s;
        size_t gb = ((size_t)(b * N + tok0 + tid)) << 6;
#pragma unroll
        for (int d = 0; d < 64; ++d) qs_g[gb + d] = (bf16)((float)sbuf[tid][d] * inv);
      }
      __syncthreads();
    } else if (proj == 1) {
      // k: exp (softmax over N is global; logits are tiny so no max shift)
      for (int i = tid; i < TOK1 * 64; i += 128) {
        int tr = i >> 6, d = i & 63;
        sbuf[tr][d] = (bf16)__expf((float)sbuf[tr][d]);
      }
      __syncthreads();
      if (tid < 64) {  // per-channel partial sum
        float s = 0.f;
#pragma unroll
        for (int tr = 0; tr < TOK1; ++tr) s += (float)sbuf[tr][tid];
        atomicAdd(&ksum[b * 64 + tid], s);
      }
      __syncthreads();
    }
  }
  __syncthreads();

  // ---- partial context: ctx[ck][cv] += sum_tok kexp[tok][ck] * v[tok][cv] ----
  for (int c = wv; c < 16; c += 4) {
    int mi = c >> 2, ni = c & 3;
    v8f acc = acc_zero();
#pragma unroll
    for (int kt = 0; kt < 2; ++kt) {
      v16bf a  = frag_ld_t(&sbuf[0][0], DPAD, mi * 16, kt * 32);  // kexp^T
      v16bf bb = frag_ld_t(&vbuf[0][0], DPAD, ni * 16, kt * 32);  // v col-major
      acc = wmma_bf16(a, bb, acc);
    }
    int col = ni * 16 + (lane & 15);
    int rb  = mi * 16 + ((lane >> 4) << 3);
#pragma unroll
    for (int r = 0; r < 8; ++r)
      atomicAdd(&ctx[((size_t)b << 12) + (size_t)(rb + r) * 64 + col], acc[r]);
  }
}

// ---------------------------------------------------------------------------
// K2: att = q @ (ctx/ksum), reproject (Wr), residual tx, LN2, fc1 (W1) -> h
// block = 128 threads (4 waves), 64 tokens. 60 KB LDS, phase-overlapped.
// ---------------------------------------------------------------------------
__global__ __launch_bounds__(128) void k2_att_ffn1(
    const float* __restrict__ x, const bf16* __restrict__ qs_g,
    const float* __restrict__ ksum, const float* __restrict__ ctx,
    const bf16* __restrict__ wr_b, const float* __restrict__ br,
    const float* __restrict__ g2, const float* __restrict__ b2ln,
    const bf16* __restrict__ w1_b, const float* __restrict__ b1,
    float* __restrict__ txg, bf16* __restrict__ h_g, int N) {
  __shared__ __align__(16) unsigned char smem[61440];
  bf16*  qsL  = (bf16*)(smem + 0);        // [64][72]
  bf16*  ctxT = (bf16*)(smem + 9216);     // [64][64]  ctxT[cv][ck]
  bf16*  abuf = (bf16*)(smem + 17408);    // [64][72]
  bf16*  wrs  = (bf16*)(smem + 26624);    // [64][64]
  float* brs  = (float*)(smem + 34816);   // [64]
  float* txb  = (float*)(smem + 0);       // [64][68] (reuses qsL/ctxT)
  bf16*  n2s  = (bf16*)(smem + 17408);    // [64][72] (reuses abuf)
  bf16*  w1s  = (bf16*)(smem + 26624);    // [256][64] (reuses wrs)
  float* b1s  = (float*)(smem + 59392);   // [256]

  const int tid  = threadIdx.x;
  const int wv   = tid >> 5;
  const int lane = tid & 31;
  const int tilesPerBatch = N / 64;
  const int b    = blockIdx.x / tilesPerBatch;
  const int tok0 = (blockIdx.x % tilesPerBatch) * 64;

  // P0: async-load q tile; build normalized transposed context
  for (int i = tid; i < 512; i += 128) {  // 64 rows x 8 chunks (128B/row)
    int tr = i >> 3, c = i & 7;
    async_cp16(&qsL[tr * DPAD + c * 8],
               qs_g + ((size_t)(b * N + tok0 + tr) << 6) + c * 8);
  }
  for (int i = tid; i < 4096; i += 128) {
    int ck = i >> 6, cv = i & 63;
    ctxT[cv * 64 + ck] = (bf16)(ctx[((size_t)b << 12) + i] / ksum[b * 64 + ck]);
  }
  async_wait0();
  __syncthreads();

  // P1: att = q @ ctxn  -> abuf
  const int mi = wv;
#pragma unroll
  for (int nt = 0; nt < 4; ++nt) {
    v8f acc = acc_zero();
#pragma unroll
    for (int kt = 0; kt < 2; ++kt) {
      v16bf a  = frag_ld(qsL, DPAD, mi * 16, kt * 32);
      v16bf bb = frag_ld(ctxT, 64, nt * 16, kt * 32);
      acc = wmma_bf16(a, bb, acc);
    }
    int col = nt * 16 + (lane & 15);
    int rb  = mi * 16 + ((lane >> 4) << 3);
#pragma unroll
    for (int r = 0; r < 8; ++r) abuf[(size_t)(rb + r) * DPAD + col] = (bf16)acc[r];
  }
  __syncthreads();

  // P2: attn_out = att @ Wr^T + br; tx = x + attn_out
  for (int i = tid; i < 512; i += 128) async_cp16(wrs + i * 8, wr_b + i * 8);
  if (tid < 64) brs[tid] = br[tid];
  async_wait0();
  __syncthreads();
#pragma unroll
  for (int nt = 0; nt < 4; ++nt) {
    v8f acc = acc_zero();
#pragma unroll
    for (int kt = 0; kt < 2; ++kt) {
      v16bf a  = frag_ld(abuf, DPAD, mi * 16, kt * 32);
      v16bf bb = frag_ld(wrs, 64, nt * 16, kt * 32);
      acc = wmma_bf16(a, bb, acc);
    }
    int col = nt * 16 + (lane & 15);
    int rb  = mi * 16 + ((lane >> 4) << 3);
#pragma unroll
    for (int r = 0; r < 8; ++r) {
      size_t gi = ((size_t)(b * N + tok0 + rb + r) << 6) + col;
      float tv = acc[r] + brs[col] + x[gi];
      txg[gi] = tv;
      txb[(rb + r) * 68 + col] = tv;
    }
  }
  __syncthreads();

  // P3: LN2
  if (tid < 64) {
    float mu = 0.f;
#pragma unroll
    for (int d = 0; d < 64; ++d) mu += txb[tid * 68 + d];
    mu *= (1.0f / 64.0f);
    float var = 0.f;
#pragma unroll
    for (int d = 0; d < 64; ++d) { float t = txb[tid * 68 + d] - mu; var += t * t; }
    var *= (1.0f / 64.0f);
    float inv = rsqrtf(var + 1e-5f);
#pragma unroll
    for (int d = 0; d < 64; ++d)
      n2s[tid * DPAD + d] = (bf16)((txb[tid * 68 + d] - mu) * inv * g2[d] + b2ln[d]);
  }
  __syncthreads();

  // P4: fc1, h = n2 @ W1^T + b1  (64 -> 256), write bf16 h
  for (int i = tid; i < 2048; i += 128) async_cp16(w1s + i * 8, w1_b + i * 8);
  for (int i = tid; i < 256; i += 128) b1s[i] = b1[i];
  async_wait0();
  __syncthreads();
  for (int c = wv; c < 64; c += 4) {
    int mi2 = c >> 4, ni = c & 15;
    v8f acc = acc_zero();
#pragma unroll
    for (int kt = 0; kt < 2; ++kt) {
      v16bf a  = frag_ld(n2s, DPAD, mi2 * 16, kt * 32);
      v16bf bb = frag_ld(w1s, 64, ni * 16, kt * 32);
      acc = wmma_bf16(a, bb, acc);
    }
    int col = ni * 16 + (lane & 15);
    int rb  = mi2 * 16 + ((lane >> 4) << 3);
#pragma unroll
    for (int r = 0; r < 8; ++r)
      h_g[((size_t)(b * N + tok0 + rb + r) << 8) + col] = (bf16)(acc[r] + b1s[col]);
  }
}

// ---------------------------------------------------------------------------
// K3: depthwise 3x3 conv (SAME, zero pad) + exact-erf GELU, bf16 in/out.
// grid = (64 spatial tiles 16x16, 4 channel groups of 64, B), 256 threads.
// ---------------------------------------------------------------------------
__global__ __launch_bounds__(256) void k3_dwconv_gelu(
    const bf16* __restrict__ h_g, const float* __restrict__ dww,
    const float* __restrict__ dwb, bf16* __restrict__ g_g, int Hdim, int Wdim) {
  __shared__ __align__(16) bf16 hs[18 * 18 * 64];  // 40.5 KB halo tile
  __shared__ float wS[64][9];
  __shared__ float bS[64];

  const int tid = threadIdx.x;
  const int ty = blockIdx.x >> 3, tx = blockIdx.x & 7;
  const int y0 = ty * 16, x0 = tx * 16;
  const int cg = blockIdx.y;  // channel group (64 ch)
  const int b  = blockIdx.z;
  const int N  = Hdim * Wdim;

  // Halo tile: 324 positions, each 64 bf16 = 128B = 8 x 16B chunks.
  for (int p = tid; p < 18 * 18; p += 256) {
    int py = p / 18, px = p % 18;
    int gy = y0 + py - 1, gx = x0 + px - 1;
    if (gy >= 0 && gy < Hdim && gx >= 0 && gx < Wdim) {
      const bf16* src = h_g + ((size_t)(b * N + gy * Wdim + gx) << 8) + cg * 64;
#pragma unroll
      for (int c8 = 0; c8 < 8; ++c8) async_cp16(&hs[p * 64 + c8 * 8], src + c8 * 8);
    } else {
      float4 z = make_float4(0.f, 0.f, 0.f, 0.f);
#pragma unroll
      for (int c8 = 0; c8 < 8; ++c8) ((float4*)&hs[p * 64])[c8] = z;
    }
  }
  for (int i = tid; i < 64 * 9; i += 256)
    wS[i / 9][i % 9] = dww[(cg * 64 + i / 9) * 9 + i % 9];
  if (tid < 64) bS[tid] = dwb[cg * 64 + tid];
  async_wait0();
  __syncthreads();

  const int sy = tid >> 4, sx = tid & 15;
  const size_t tokbase = ((size_t)(b * N + (y0 + sy) * Wdim + (x0 + sx)) << 8) + cg * 64;
#pragma unroll 4
  for (int c = 0; c < 64; ++c) {
    float s = bS[c];
#pragma unroll
    for (int dy = 0; dy < 3; ++dy)
#pragma unroll
      for (int dx = 0; dx < 3; ++dx)
        s += (float)hs[((sy + dy) * 18 + (sx + dx)) * 64 + c] * wS[c][dy * 3 + dx];
    // exact GELU: 0.5*x*(1+erf(x/sqrt(2)))
    float gl = 0.5f * s * (1.0f + erff(s * 0.70710678118654752f));
    g_g[tokbase + c] = (bf16)gl;
  }
}

// ---------------------------------------------------------------------------
// K4: out = tx + gelu_out @ W2^T + b2   (256 -> 64), f32 output.
// block = 128 threads (4 waves), 32 tokens.
// ---------------------------------------------------------------------------
__global__ __launch_bounds__(128) void k4_ffn2(
    const bf16* __restrict__ g_g, const bf16* __restrict__ w2_b,
    const float* __restrict__ b2, const float* __restrict__ txg,
    float* __restrict__ out, int N) {
  __shared__ __align__(16) bf16 gL[32 * 264];   // 16.5 KB
  __shared__ __align__(16) bf16 w2s[64 * 256];  // 32 KB
  __shared__ float b2s[64];

  const int tid  = threadIdx.x;
  const int wv   = tid >> 5;
  const int lane = tid & 31;
  const int tilesPerBatch = N / 32;
  const int b    = blockIdx.x / tilesPerBatch;
  const int tok0 = (blockIdx.x % tilesPerBatch) * 32;

  for (int i = tid; i < 1024; i += 128) {  // 32 rows x 32 chunks (512B/row)
    int tr = i >> 5, c = i & 31;
    async_cp16(&gL[tr * 264 + c * 8],
               g_g + ((size_t)(b * N + tok0 + tr) << 8) + c * 8);
  }
  for (int i = tid; i < 2048; i += 128) async_cp16(w2s + i * 8, w2_b + i * 8);
  if (tid < 64) b2s[tid] = b2[tid];
  async_wait0();
  __syncthreads();

  for (int c = wv; c < 8; c += 4) {
    int mi = c >> 2, ni = c & 3;
    v8f acc = acc_zero();
#pragma unroll
    for (int kt = 0; kt < 8; ++kt) {
      v16bf a  = frag_ld(gL, 264, mi * 16, kt * 32);
      v16bf bb = frag_ld(w2s, 256, ni * 16, kt * 32);
      acc = wmma_bf16(a, bb, acc);
    }
    int col = ni * 16 + (lane & 15);
    int rb  = mi * 16 + ((lane >> 4) << 3);
#pragma unroll
    for (int r = 0; r < 8; ++r) {
      size_t gi = ((size_t)(b * N + tok0 + rb + r) << 6) + col;
      out[gi] = acc[r] + b2s[col] + txg[gi];
    }
  }
}

// ---------------------------------------------------------------------------
extern "C" void kernel_launch(void* const* d_in, const int* in_sizes, int n_in,
                              void* d_out, int out_size, void* d_ws, size_t ws_size,
                              hipStream_t stream) {
  const float* x     = (const float*)d_in[0];
  const float* ln1_s = (const float*)d_in[3];
  const float* ln1_b = (const float*)d_in[4];
  const float* Wk    = (const float*)d_in[5];
  const float* bk    = (const float*)d_in[6];
  const float* Wq    = (const float*)d_in[7];
  const float* bq    = (const float*)d_in[8];
  const float* Wv    = (const float*)d_in[9];
  const float* bv    = (const float*)d_in[10];
  const float* Wr    = (const float*)d_in[11];
  const float* br    = (const float*)d_in[12];
  const float* ln2_s = (const float*)d_in[13];
  const float* ln2_b = (const float*)d_in[14];
  const float* W1    = (const float*)d_in[15];
  const float* b1    = (const float*)d_in[16];
  const float* dww   = (const float*)d_in[17];
  const float* dwb   = (const float*)d_in[18];
  const float* W2    = (const float*)d_in[19];
  const float* b2    = (const float*)d_in[20];

  const int Hdim = 128, Wdim = 128;
  const int N = Hdim * Wdim;             // 16384
  const int B = in_sizes[0] / (N * 64);  // 16

  // workspace layout
  char* ws = (char*)d_ws;
  size_t off = 0;
  bf16*  qs_g = (bf16*)(ws + off);  off += (size_t)B * N * 64 * 2;   // 32 MB
  float* txg  = (float*)(ws + off); off += (size_t)B * N * 64 * 4;   // 64 MB
  bf16*  h_g  = (bf16*)(ws + off);  off += (size_t)B * N * 256 * 2;  // 128 MB
  bf16*  g_g  = (bf16*)(ws + off);  off += (size_t)B * N * 256 * 2;  // 128 MB
  float* ksum = (float*)(ws + off); off += (size_t)B * 64 * 4;
  float* ctx  = (float*)(ws + off); off += (size_t)B * 64 * 64 * 4;
  bf16*  wq_b = (bf16*)(ws + off);  off += 4096 * 2;
  bf16*  wk_b = (bf16*)(ws + off);  off += 4096 * 2;
  bf16*  wv_b = (bf16*)(ws + off);  off += 4096 * 2;
  bf16*  wr_b = (bf16*)(ws + off);  off += 4096 * 2;
  bf16*  w1_b = (bf16*)(ws + off);  off += 16384 * 2;
  bf16*  w2_b = (bf16*)(ws + off);  off += 16384 * 2;

  // K0: zero accumulators + convert weights to bf16 once
  {
    int nk = B * 64, nc = B * 64 * 64;
    int work = nc > 16384 ? nc : 16384;
    k0_prep<<<(work + 255) / 256, 256, 0, stream>>>(Wq, Wk, Wv, Wr, W1, W2, wq_b,
                                                    wk_b, wv_b, wr_b, w1_b, w2_b,
                                                    ksum, ctx, nk, nc);
  }
  // K1
  k1_proj<<<B * (N / TOK1), 128, 0, stream>>>(x, ln1_s, ln1_b, wq_b, bq, wk_b, bk,
                                              wv_b, bv, qs_g, ksum, ctx, N);
  // K2
  k2_att_ffn1<<<B * (N / 64), 128, 0, stream>>>(x, qs_g, ksum, ctx, wr_b, br, ln2_s,
                                                ln2_b, w1_b, b1, txg, h_g, N);
  // K3
  {
    dim3 grid((Hdim / 16) * (Wdim / 16), 4, B);
    k3_dwconv_gelu<<<grid, 256, 0, stream>>>(h_g, dww, dwb, g_g, Hdim, Wdim);
  }
  // K4
  k4_ffn2<<<B * (N / 32), 128, 0, stream>>>(g_g, w2_b, b2, txg, (float*)d_out, N);
}
